// NTXentLoss_21758304322066
// MI455X (gfx1250) — matrix operand order repaired and checked by reading, not compiled, not run on hardware
//
#include <hip/hip_runtime.h>
#include <hip/hip_fp16.h>

typedef __attribute__((ext_vector_type(16))) _Float16 v16h;
typedef __attribute__((ext_vector_type(8)))  _Float16 v8h;
typedef __attribute__((ext_vector_type(4)))  _Float16 v4h;
typedef __attribute__((ext_vector_type(8)))  float    v8f;

#define BDIM   8192
#define DDIM   128
#define NCHUNK 16
#define TILES_PER_CHUNK 32   /* 512 column tiles / 16 chunks */

// ---------------------------------------------------------------------------
// prep: per-row L2 norm, fold cosine scale (and 1/T into zjs), split fp32 ->
// (f16 hi, f16 lo). One wave32 per row; lane handles 4 consecutive elements.
// ---------------------------------------------------------------------------
__global__ __launch_bounds__(256) void ntxent_prep(
    const float* __restrict__ zis, const float* __restrict__ zjs,
    _Float16* __restrict__ Ahi, _Float16* __restrict__ Alo,
    _Float16* __restrict__ Bhi, _Float16* __restrict__ Blo)
{
  const int gwave = (int)((blockIdx.x * blockDim.x + threadIdx.x) >> 5);
  const int lane  = (int)(threadIdx.x & 31);
  const bool isB  = gwave >= BDIM;
  const int  row  = isB ? (gwave - BDIM) : gwave;

  const float* src = (isB ? zjs : zis) + (size_t)row * DDIM;
  const float4 x = ((const float4*)src)[lane];
  float ss = x.x * x.x + x.y * x.y + x.z * x.z + x.w * x.w;
#pragma unroll
  for (int off = 1; off < 32; off <<= 1) ss += __shfl_xor(ss, off, 32);

  // zis rows scaled by 1/||zi||; zjs rows by (1/||zj||)*(1/T) with T=0.5
  const float scale = (isB ? 2.0f : 1.0f) / sqrtf(ss);
  const float y0 = x.x * scale, y1 = x.y * scale, y2 = x.z * scale, y3 = x.w * scale;

  const _Float16 h0 = (_Float16)y0, h1 = (_Float16)y1,
                 h2 = (_Float16)y2, h3 = (_Float16)y3;
  const _Float16 l0 = (_Float16)(y0 - (float)h0), l1 = (_Float16)(y1 - (float)h1),
                 l2 = (_Float16)(y2 - (float)h2), l3 = (_Float16)(y3 - (float)h3);

  _Float16* dhi = (isB ? Bhi : Ahi) + (size_t)row * DDIM + lane * 4;
  _Float16* dlo = (isB ? Blo : Alo) + (size_t)row * DDIM + lane * 4;
  *(v4h*)dhi = (v4h){h0, h1, h2, h3};
  *(v4h*)dlo = (v4h){l0, l1, l2, l3};
}

// ---------------------------------------------------------------------------
// main: one wave per (16-row strip, 32-tile column chunk).
// Computes scaled-sim tiles with split-f16 WMMA (hi*hi + hi*lo + lo*hi),
// accumulates sum-of-exp per C slot (no max needed: |logit| <= 2), handles
// the G=16 mask block (== one WMMA tile) by keeping only the diagonal.
// ---------------------------------------------------------------------------
__global__ __launch_bounds__(256) void ntxent_main(
    const _Float16* __restrict__ Ahi, const _Float16* __restrict__ Alo,
    const _Float16* __restrict__ Bhi, const _Float16* __restrict__ Blo,
    float* __restrict__ lpart, float* __restrict__ pos)
{
  const int lane  = (int)(threadIdx.x & 31);
  const int gw    = (int)((blockIdx.x * blockDim.x + threadIdx.x) >> 5);
  const int strip = gw >> 4;            // 0..511 row tile
  const int chunk = gw & (NCHUNK - 1);  // 0..15 column chunk
  const int kh    = lane >> 4;          // half-wave selector
  const int ln    = lane & 15;

  // A-matrix (16x32 f16) per-lane layout: lanes 0-15 row M=lane, contiguous
  // K=0..15; lanes 16-31 same rows, K=16..31. Preload full
  // K=128 (hi+lo) once; reused across all 32 column tiles.
  union { v16h v; v8h h[2]; } ahi[4], alo[4];
  {
    const _Float16* ar_hi = Ahi + (size_t)(strip * 16 + ln) * DDIM;
    const _Float16* ar_lo = Alo + (size_t)(strip * 16 + ln) * DDIM;
#pragma unroll
    for (int kt = 0; kt < 4; ++kt) {
      ahi[kt].h[0] = *(const v8h*)(ar_hi + kt * 32 + 16 * kh);
      ahi[kt].h[1] = *(const v8h*)(ar_hi + kt * 32 + 16 * kh + 8);
      alo[kt].h[0] = *(const v8h*)(ar_lo + kt * 32 + 16 * kh);
      alo[kt].h[1] = *(const v8h*)(ar_lo + kt * 32 + 16 * kh + 8);
    }
  }

  float lsum[8], posv[8];
#pragma unroll
  for (int v = 0; v < 8; ++v) { lsum[v] = 0.f; posv[v] = 0.f; }

  const int c0 = chunk * TILES_PER_CHUNK;
  for (int c = c0; c < c0 + TILES_PER_CHUNK; ++c) {
    // B-matrix (32x16 f16): lane holds column N=ln, contiguous K range
    // [16*kh, 16*kh+16) -> one 32-byte contiguous load per K-step.
    const _Float16* bh = Bhi + (size_t)(c * 16 + ln) * DDIM + 16 * kh;
    const _Float16* bl = Blo + (size_t)(c * 16 + ln) * DDIM + 16 * kh;
    v16h bhv[4], blv[4];
#pragma unroll
    for (int kt = 0; kt < 4; ++kt) {
      bhv[kt] = *(const v16h*)(bh + kt * 32);
      blv[kt] = *(const v16h*)(bl + kt * 32);
    }

    v8f a0 = {}, a1 = {}, a2 = {};   // hi*hi, hi*lo, lo*hi (independent chains)
#pragma unroll
    for (int kt = 0; kt < 4; ++kt) {
      a0 = __builtin_amdgcn_wmma_f32_16x16x32_f16(false, ahi[kt].v, false, bhv[kt],
                                                  (short)0, a0, false, false);
      a1 = __builtin_amdgcn_wmma_f32_16x16x32_f16(false, ahi[kt].v, false, blv[kt],
                                                  (short)0, a1, false, false);
      a2 = __builtin_amdgcn_wmma_f32_16x16x32_f16(false, alo[kt].v, false, bhv[kt],
                                                  (short)0, a2, false, false);
    }

    // C/D slot (VGPR v, lane): row M = v + 8*kh, col N = ln.
    if (c == strip) {
      // own mask block: only the true diagonal (the positive) contributes
#pragma unroll
      for (int v = 0; v < 8; ++v) {
        const float s = (a0[v] + a1[v]) + a2[v];
        if (ln == v + 8 * kh) { posv[v] = s; lsum[v] += __expf(s); }
      }
    } else {
#pragma unroll
      for (int v = 0; v < 8; ++v) {
        const float s = (a0[v] + a1[v]) + a2[v];
        lsum[v] += __expf(s);
      }
    }
  }

  // Reduce across the 16 lanes sharing each C row; write chunk partials.
#pragma unroll
  for (int v = 0; v < 8; ++v) {
    float l = lsum[v], p = posv[v];
#pragma unroll
    for (int off = 1; off < 16; off <<= 1) {
      l += __shfl_xor(l, off, 32);
      p += __shfl_xor(p, off, 32);
    }
    if (ln == 0) {
      const int row = strip * 16 + v + 8 * kh;
      lpart[chunk * BDIM + row] = l;
      if (strip >= c0 && strip < c0 + TILES_PER_CHUNK) pos[row] = p;
    }
  }
}

// ---------------------------------------------------------------------------
// finalize: deterministic merge of chunk partials + scalar reduction.
// loss_i = log(sum_exp_i) - pos_i ;  out = mean(loss)
// ---------------------------------------------------------------------------
__global__ __launch_bounds__(256) void ntxent_finalize(
    const float* __restrict__ lpart, const float* __restrict__ pos,
    float* __restrict__ out)
{
  __shared__ float sm[256];
  const int tid = (int)threadIdx.x;
  float acc = 0.f;
  for (int row = tid; row < BDIM; row += 256) {
    float l = 0.f;
#pragma unroll
    for (int c = 0; c < NCHUNK; ++c) l += lpart[c * BDIM + row];
    acc += logf(l) - pos[row];
  }
  sm[tid] = acc;
  __syncthreads();
  for (int s = 128; s > 0; s >>= 1) {
    if (tid < s) sm[tid] += sm[tid + s];
    __syncthreads();
  }
  if (tid == 0) out[0] = sm[0] / (float)BDIM;
}

// ---------------------------------------------------------------------------
extern "C" void kernel_launch(void* const* d_in, const int* in_sizes, int n_in,
                              void* d_out, int out_size, void* d_ws, size_t ws_size,
                              hipStream_t stream) {
  (void)in_sizes; (void)n_in; (void)out_size; (void)ws_size;
  const float* zis = (const float*)d_in[0];
  const float* zjs = (const float*)d_in[1];
  float* out = (float*)d_out;

  char* w = (char*)d_ws;
  const size_t MAT = (size_t)BDIM * DDIM * sizeof(_Float16);  // 2 MB each
  _Float16* Ahi = (_Float16*)(w + 0 * MAT);
  _Float16* Alo = (_Float16*)(w + 1 * MAT);
  _Float16* Bhi = (_Float16*)(w + 2 * MAT);
  _Float16* Blo = (_Float16*)(w + 3 * MAT);
  float* lpart  = (float*)(w + 4 * MAT);                               // 512 KB
  float* pos    = (float*)(w + 4 * MAT + (size_t)NCHUNK * BDIM * 4);   // 32 KB

  // 1) norms + scale-fold + f16 hi/lo split: 2*8192 waves, 8 waves/block
  ntxent_prep<<<(2 * BDIM) / 8, 256, 0, stream>>>(zis, zjs, Ahi, Alo, Bhi, Blo);

  // 2) fused GEMM + masked sum-exp: 512 strips * 16 chunks = 8192 waves
  ntxent_main<<<(512 * NCHUNK) / 8, 256, 0, stream>>>(Ahi, Alo, Bhi, Blo, lpart, pos);

  // 3) deterministic final reduction to the scalar loss
  ntxent_finalize<<<1, 256, 0, stream>>>(lpart, pos, out);
}
